// Rigid_MultiHeadAttention_56925496541567
// MI455X (gfx1250) — compile-verified, matrix operand order checked by reading, not run on hardware
//
#include <hip/hip_runtime.h>
#include <hip/hip_bf16.h>

// Problem constants (from reference)
#define B_   4
#define L_   512
#define DM_  512
#define H_   8
#define HD_  64
#define ML_  (B_ * L_)   // 2048 total rows
#define BH_  (B_ * H_)   // 32

typedef __attribute__((ext_vector_type(16))) _Float16 v16h;
typedef __attribute__((ext_vector_type(8)))  float    v8f;

// ---------------------------------------------------------------- helpers ---

__device__ __forceinline__ v8f wmma_f16(v16h a, v16h b, v8f c) {
  // v_wmma_f32_16x16x32_f16: D = A(16x32 f16) * B(32x16 f16) + C(16x16 f32)
  return __builtin_amdgcn_wmma_f32_16x16x32_f16(
      /*neg_a=*/false, a, /*neg_b=*/false, b,
      /*c_mod=*/(short)0, c, /*reuse_a=*/false, /*reuse_b=*/false);
}

__device__ __forceinline__ v8f zero8() {
  v8f z;
#pragma unroll
  for (int i = 0; i < 8; ++i) z[i] = 0.0f;
  return z;
}

// CDNA5 16-bit A/B fragment per-lane layout:
//   lanes 0-15 : K = {0..7} in halves 0..7, K = {16..23} in halves 8..15
//   lanes 16-31: K = {8..15} and {24..31}
// Caller passes p already offset by (k0 + ((lane&16)?8:0)); the two 8-half
// chunks are then p[0..7] and p[16..23] — contiguous 16B/32B runs.
__device__ __forceinline__ v16h frag_f32(const float* __restrict__ p) {
  v16h f;
#pragma unroll
  for (int i = 0; i < 8; ++i) {
    f[i]     = (_Float16)p[i];
    f[i + 8] = (_Float16)p[i + 16];
  }
  return f;
}

__device__ __forceinline__ v16h frag_f16(const _Float16* __restrict__ p) {
  v16h f;
#pragma unroll
  for (int i = 0; i < 8; ++i) {
    f[i]     = p[i];
    f[i + 8] = p[i + 16];
  }
  return f;
}

__device__ __forceinline__ float wave_sum(float v) {
#pragma unroll
  for (int o = 16; o > 0; o >>= 1) v += __shfl_xor(v, o, 32);
  return v;
}
__device__ __forceinline__ float wave_max(float v) {
#pragma unroll
  for (int o = 16; o > 0; o >>= 1) v = fmaxf(v, __shfl_xor(v, o, 32));
  return v;
}

// ------------------------------------------------------------ K0: QKV proj --
// Y = X @ W^T + b  for W in {Wq,Wk,Wv}.  One wave per 16x64 output tile.
// Q,K stored [B,H,L,HD] f16 (rows contiguous over HD); V stored transposed
// [B,H,HD,L] f16 so attn_v B-fragments are contiguous.
__global__ void qkv_proj_kernel(const float* __restrict__ X,
                                const float* __restrict__ Wq, const float* __restrict__ bq,
                                const float* __restrict__ Wk, const float* __restrict__ bk,
                                const float* __restrict__ Wv, const float* __restrict__ bv,
                                _Float16* __restrict__ qh,
                                _Float16* __restrict__ kh,
                                _Float16* __restrict__ vt) {
  const int lane  = threadIdx.x & 31;
  const int mBase = blockIdx.x * 16;   // 0..2047 rows, 128 tiles
  const int nBase = blockIdx.y * 64;   // 0..511 cols, 8 strips
  const int mat   = blockIdx.z;        // 0=Q 1=K 2=V

  const float* W;
  const float* bias;
  if (mat == 0)      { W = Wq; bias = bq; }
  else if (mat == 1) { W = Wk; bias = bk; }
  else               { W = Wv; bias = bv; }

  const int aRow  = mBase + (lane & 15);
  const int koffs = (lane & 16) ? 8 : 0;

  v8f acc[4];
#pragma unroll
  for (int t = 0; t < 4; ++t) acc[t] = zero8();

  for (int k0 = 0; k0 < DM_; k0 += 32) {
    v16h a = frag_f32(X + aRow * DM_ + k0 + koffs);
#pragma unroll
    for (int t = 0; t < 4; ++t) {
      const int col = nBase + t * 16 + (lane & 15);   // output channel n
      v16h bfr = frag_f32(W + col * DM_ + k0 + koffs); // B[k][n] = W[n][k]
      acc[t] = wmma_f16(a, bfr, acc[t]);
    }
  }

  // D layout: element (M = r + 8*(lane>=16), N = lane&15) in acc[t][r]
  const int mOff = (lane >> 4) << 3;
#pragma unroll
  for (int t = 0; t < 4; ++t) {
    const int n = nBase + t * 16 + (lane & 15);
    const int h = n >> 6, d = n & 63;
#pragma unroll
    for (int r = 0; r < 8; ++r) {
      const int m  = mBase + r + mOff;
      const int bb = m >> 9, ll = m & 511;
      const float y = acc[t][r] + bias[n];
      if (mat == 0)
        qh[(((bb * H_ + h) * L_) + ll) * HD_ + d] = (_Float16)y;
      else if (mat == 1)
        kh[(((bb * H_ + h) * L_) + ll) * HD_ + d] = (_Float16)y;
      else
        vt[(((bb * H_ + h) * HD_) + d) * L_ + ll] = (_Float16)y;
    }
  }
}

// ------------------------------------------------------------ K1: v3d proj --
__global__ void v3d_proj_kernel(const float* __restrict__ X,
                                const float* __restrict__ Wv3,
                                const float* __restrict__ bv3,
                                float* __restrict__ v3d) {
  const int t = blockIdx.x * blockDim.x + threadIdx.x;
  if (t >= ML_ * 3) return;
  const int row = t / 3, n = t % 3;
  const float* xp = X + row * DM_;
  const float* wp = Wv3 + n * DM_;
  float s = bv3[n];
#pragma unroll 8
  for (int k = 0; k < DM_; ++k) s = fmaf(xp[k], wp[k], s);
  v3d[row * 3 + n] = s;
}

// ------------------------------------------------- K2: scores = QK^T + bias --
// One wave per 16x64 tile of a (b,h) 512x512 score matrix.  Epilogue fuses
// 1/sqrt(HD), the RBF-MLP distance bias, and the attention mask.
__global__ void scores_kernel(const _Float16* __restrict__ qh,
                              const _Float16* __restrict__ kh,
                              const float* __restrict__ dist,
                              const int*   __restrict__ msk,
                              const float* __restrict__ Wm,
                              const float* __restrict__ bm,
                              float* __restrict__ sc) {
  const int lane  = threadIdx.x & 31;
  const int mBase = blockIdx.x * 16;   // query tile (32)
  const int nBase = blockIdx.y * 64;   // key strip (8)
  const int bh    = blockIdx.z;        // 32
  const int b     = bh >> 3;

  const int koffs = (lane & 16) ? 8 : 0;
  const _Float16* Qrow  = qh + (bh * L_ + mBase + (lane & 15)) * HD_;
  const _Float16* Kbase = kh + bh * L_ * HD_;

  v8f acc[4];
#pragma unroll
  for (int t = 0; t < 4; ++t) acc[t] = zero8();

#pragma unroll
  for (int k0 = 0; k0 < HD_; k0 += 32) {
    v16h a = frag_f16(Qrow + k0 + koffs);
#pragma unroll
    for (int t = 0; t < 4; ++t) {
      const int col = nBase + t * 16 + (lane & 15);
      v16h bfr = frag_f16(Kbase + col * HD_ + k0 + koffs); // B[k][n]=K[n][k]
      acc[t] = wmma_f16(a, bfr, acc[t]);
    }
  }

  float wr[16];
#pragma unroll
  for (int r = 0; r < 16; ++r) wr[r] = Wm[r];
  const float bml = bm[0];

  const int mOff = (lane >> 4) << 3;
#pragma unroll
  for (int t = 0; t < 4; ++t) {
    const int j = nBase + t * 16 + (lane & 15);
#pragma unroll
    for (int r = 0; r < 8; ++r) {
      const int i  = mBase + r + mOff;
      const int ij = (b * L_ + i) * L_ + j;
      float s = acc[t][r] * 0.125f + bml;          // 1/sqrt(64)
      const float d = dist[ij];
#pragma unroll
      for (int rr = 0; rr < 16; ++rr) {
        // mu_r = r*20/15 ; sigma = 20/16 -> 1/sigma = 0.8
        const float u = (d - (float)rr * (20.0f / 15.0f)) * 0.8f;
        s = fmaf(__expf(-u * u), wr[rr], s);
      }
      if (msk[ij] == 0) s = -1e9f;
      sc[(bh * L_ + i) * L_ + j] = s;
    }
  }
}

// ------------------------------------------------ K3: per-head row softmax --
__global__ void softmax_head_kernel(const float* __restrict__ sc,
                                    _Float16* __restrict__ aw) {
  const int lane = threadIdx.x & 31;
  const int wid  = threadIdx.x >> 5;
  const int row  = blockIdx.x * 8 + wid;       // 0 .. B*H*L-1 (16384)
  const float* sp = sc + (size_t)row * L_;

  float v[16];
  float mx = -3.4e38f;
#pragma unroll
  for (int t = 0; t < 16; ++t) { v[t] = sp[lane + 32 * t]; mx = fmaxf(mx, v[t]); }
  mx = wave_max(mx);
  float sum = 0.0f;
#pragma unroll
  for (int t = 0; t < 16; ++t) { v[t] = __expf(v[t] - mx); sum += v[t]; }
  sum = wave_sum(sum);
  const float inv = 1.0f / sum;
  _Float16* op = aw + (size_t)row * L_;
#pragma unroll
  for (int t = 0; t < 16; ++t) op[lane + 32 * t] = (_Float16)(v[t] * inv);
}

// -------------------------------------- K4: head-sum + merged row softmax --
__global__ void softmax_merged_kernel(const float* __restrict__ sc,
                                      float* __restrict__ wm) {
  const int lane = threadIdx.x & 31;
  const int wid  = threadIdx.x >> 5;
  const int row  = blockIdx.x * 8 + wid;       // 0 .. B*L-1 (2048)
  const int b = row >> 9, i = row & 511;
  const float* base = sc + ((size_t)(b * H_) * L_ + i) * L_;  // head stride L*L

  float v[16];
  float mx = -3.4e38f;
#pragma unroll
  for (int t = 0; t < 16; ++t) {
    const int j = lane + 32 * t;
    float s = 0.0f;
#pragma unroll
    for (int h = 0; h < H_; ++h) s += base[(size_t)h * L_ * L_ + j];
    v[t] = s;
    mx = fmaxf(mx, s);
  }
  mx = wave_max(mx);
  float sum = 0.0f;
#pragma unroll
  for (int t = 0; t < 16; ++t) { v[t] = __expf(v[t] - mx); sum += v[t]; }
  sum = wave_sum(sum);
  const float inv = 1.0f / sum;
  float* op = wm + (size_t)row * L_;
#pragma unroll
  for (int t = 0; t < 16; ++t) op[lane + 32 * t] = v[t] * inv;
}

// -------------------------------------------------- K5: attn_v = AW @ V ----
// One wave computes a 16x64 tile of the [L, HD] result for one (b,h);
// written straight into [B, L, DM] order for the final FC.
__global__ void attnv_kernel(const _Float16* __restrict__ aw,
                             const _Float16* __restrict__ vt,
                             float* __restrict__ cv) {
  const int lane  = threadIdx.x & 31;
  const int mBase = blockIdx.x * 16;   // 32 tiles
  const int bh    = blockIdx.y;        // 32
  const int b = bh >> 3, h = bh & 7;

  const int koffs = (lane & 16) ? 8 : 0;
  const _Float16* Arow  = aw + ((size_t)bh * L_ + mBase + (lane & 15)) * L_;
  const _Float16* Bbase = vt + (size_t)bh * HD_ * L_;

  v8f acc[4];
#pragma unroll
  for (int t = 0; t < 4; ++t) acc[t] = zero8();

  for (int k0 = 0; k0 < L_; k0 += 32) {
    v16h a = frag_f16(Arow + k0 + koffs);
#pragma unroll
    for (int t = 0; t < 4; ++t) {
      const int col = t * 16 + (lane & 15);              // d in 0..63
      v16h bfr = frag_f16(Bbase + col * L_ + k0 + koffs); // B[k][d]=Vt[d][k]
      acc[t] = wmma_f16(a, bfr, acc[t]);
    }
  }

  const int mOff = (lane >> 4) << 3;
#pragma unroll
  for (int t = 0; t < 4; ++t) {
    const int d = t * 16 + (lane & 15);
#pragma unroll
    for (int r = 0; r < 8; ++r) {
      const int i = mBase + r + mOff;
      cv[((size_t)(b * L_) + i) * DM_ + h * HD_ + d] = acc[t][r];
    }
  }
}

// ---------------------------------------------------- K6: geometric frame --
// attn_frame[b,i,:] = mean_j wm[b,i,j] * cross(ad[b,i,j,:], v3d[b,j,:]*colsum)
// where colsum[n] = sum_m orientation[b,i,j,m,n]   (per reference einsum).
__global__ void frame_kernel(const float* __restrict__ wm,
                             const float* __restrict__ ad,
                             const float* __restrict__ ori,
                             const float* __restrict__ v3d,
                             float* __restrict__ fr) {
  const int lane = threadIdx.x & 31;
  const int wid  = threadIdx.x >> 5;
  const int row  = blockIdx.x * 8 + wid;       // 0 .. B*L-1 (2048)
  const int b = row >> 9, i = row & 511;

  float a0 = 0.0f, a1 = 0.0f, a2 = 0.0f;
  for (int j = lane; j < L_; j += 32) {
    const size_t ij = ((size_t)(b * L_) + i) * L_ + j;
    const float w = wm[ij];
    const float* op = ori + ij * 9;
    const float* dp = ad + ij * 3;
    const float* vp = v3d + (size_t)(b * L_ + j) * 3;
    __builtin_prefetch(op + 32 * 9, 0, 1);     // global_prefetch_b8 on the big stream
    const float c0 = op[0] + op[3] + op[6];
    const float c1 = op[1] + op[4] + op[7];
    const float c2 = op[2] + op[5] + op[8];
    const float m0 = vp[0] * c0, m1 = vp[1] * c1, m2 = vp[2] * c2;
    const float d0 = dp[0], d1 = dp[1], d2 = dp[2];
    a0 = fmaf(w, d1 * m2 - d2 * m1, a0);
    a1 = fmaf(w, d2 * m0 - d0 * m2, a1);
    a2 = fmaf(w, d0 * m1 - d1 * m0, a2);
  }
  a0 = wave_sum(a0); a1 = wave_sum(a1); a2 = wave_sum(a2);
  if (lane == 0) {
    const float s = 1.0f / (float)L_;          // mean over j
    fr[row * 3 + 0] = a0 * s;
    fr[row * 3 + 1] = a1 * s;
    fr[row * 3 + 2] = a2 * s;
  }
}

// ------------------------------------------- K7: out = concat @ Wfc^T + b --
// WMMA handles the K=512 attn_v part; the 3 frame columns + bias are a
// scalar epilogue (Wfc is [DM, DM+3], row-major length 515).
__global__ void fc_kernel(const float* __restrict__ cv,
                          const float* __restrict__ fr,
                          const float* __restrict__ Wfc,
                          const float* __restrict__ bfc,
                          float* __restrict__ out) {
  const int lane  = threadIdx.x & 31;
  const int mBase = blockIdx.x * 16;   // 128 tiles of 2048 rows
  const int nBase = blockIdx.y * 64;   // 8 strips of 512 cols
  const int koffs = (lane & 16) ? 8 : 0;
  const int LDW   = DM_ + 3;           // 515

  const float* Arow = cv + (size_t)(mBase + (lane & 15)) * DM_;

  v8f acc[4];
#pragma unroll
  for (int t = 0; t < 4; ++t) acc[t] = zero8();

  for (int k0 = 0; k0 < DM_; k0 += 32) {
    v16h a = frag_f32(Arow + k0 + koffs);
#pragma unroll
    for (int t = 0; t < 4; ++t) {
      const int col = nBase + t * 16 + (lane & 15);
      v16h bfr = frag_f32(Wfc + (size_t)col * LDW + k0 + koffs);
      acc[t] = wmma_f16(a, bfr, acc[t]);
    }
  }

  const int mOff = (lane >> 4) << 3;
#pragma unroll
  for (int t = 0; t < 4; ++t) {
    const int n = nBase + t * 16 + (lane & 15);
    const float* wrow = Wfc + (size_t)n * LDW;
    const float w512 = wrow[512], w513 = wrow[513], w514 = wrow[514];
    const float bn = bfc[n];
#pragma unroll
    for (int r = 0; r < 8; ++r) {
      const int m = mBase + r + mOff;
      float y = acc[t][r] + bn;
      y = fmaf(fr[m * 3 + 0], w512, y);
      y = fmaf(fr[m * 3 + 1], w513, y);
      y = fmaf(fr[m * 3 + 2], w514, y);
      out[(size_t)m * DM_ + n] = y;
    }
  }
}

// ------------------------------------------------------------------ launch --
extern "C" void kernel_launch(void* const* d_in, const int* in_sizes, int n_in,
                              void* d_out, int out_size, void* d_ws, size_t ws_size,
                              hipStream_t stream) {
  const float* x    = (const float*)d_in[0];
  const float* ad   = (const float*)d_in[1];
  const float* ori  = (const float*)d_in[2];
  const int*   msk  = (const int*)  d_in[3];
  const float* dist = (const float*)d_in[4];
  const float* Wq   = (const float*)d_in[5];
  const float* bq   = (const float*)d_in[6];
  const float* Wk   = (const float*)d_in[7];
  const float* bk   = (const float*)d_in[8];
  const float* Wv   = (const float*)d_in[9];
  const float* bv   = (const float*)d_in[10];
  const float* Wv3  = (const float*)d_in[11];
  const float* bv3  = (const float*)d_in[12];
  const float* Wfc  = (const float*)d_in[13];
  const float* bfc  = (const float*)d_in[14];
  const float* Wm   = (const float*)d_in[15];
  const float* bm   = (const float*)d_in[16];

  char* ws = (char*)d_ws;
  // Workspace map (total ~63 MiB + 48 KiB):
  _Float16* qh  = (_Float16*)(ws + ((size_t)0  << 20));  // 2 MiB  [B,H,L,HD] f16
  _Float16* kh  = (_Float16*)(ws + ((size_t)2  << 20));  // 2 MiB
  _Float16* vt  = (_Float16*)(ws + ((size_t)4  << 20));  // 2 MiB  [B,H,HD,L] f16
  float*    v3d = (float*)   (ws + ((size_t)6  << 20));  // 24 KiB [B,L,3]
  float*    sc  = (float*)   (ws + ((size_t)7  << 20));  // 32 MiB [B,H,L,L] f32
  _Float16* aw  = (_Float16*)(ws + ((size_t)39 << 20));  // 16 MiB [B,H,L,L] f16
  float*    wmw = (float*)   (ws + ((size_t)55 << 20));  // 4 MiB  [B,L,L]
  float*    cv  = (float*)   (ws + ((size_t)59 << 20));  // 4 MiB  [B,L,DM]
  float*    fr  = (float*)   (ws + ((size_t)63 << 20));  // 24 KiB [B,L,3]
  float*    out = (float*)d_out;                         // [B,L,DM]

  // K0: Q/K/V projections (WMMA)
  qkv_proj_kernel<<<dim3(ML_ / 16, DM_ / 64, 3), 32, 0, stream>>>(
      x, Wq, bq, Wk, bk, Wv, bv, qh, kh, vt);
  // K1: v_3d projection
  v3d_proj_kernel<<<(ML_ * 3 + 255) / 256, 256, 0, stream>>>(x, Wv3, bv3, v3d);
  // K2: scores + RBF bias + mask (WMMA)
  scores_kernel<<<dim3(L_ / 16, L_ / 64, BH_), 32, 0, stream>>>(
      qh, kh, dist, msk, Wm, bm, sc);
  // K3: per-head softmax -> f16 weights
  softmax_head_kernel<<<(BH_ * L_) / 8, 256, 0, stream>>>(sc, aw);
  // K4: merged (head-summed) softmax
  softmax_merged_kernel<<<ML_ / 8, 256, 0, stream>>>(sc, wmw);
  // K5: attn_v = weights @ V (WMMA)
  attnv_kernel<<<dim3(L_ / 16, BH_), 32, 0, stream>>>(aw, vt, cv);
  // K6: geometric frame
  frame_kernel<<<ML_ / 8, 256, 0, stream>>>(wmw, ad, ori, v3d, fr);
  // K7: final FC (WMMA + frame/bias epilogue)
  fc_kernel<<<dim3(ML_ / 16, DM_ / 64), 32, 0, stream>>>(cv, fr, Wfc, bfc, out);
}
